// RNN_79379585565280
// MI455X (gfx1250) — matrix operand order repaired and checked by reading, not compiled, not run on hardware
//
#include <hip/hip_runtime.h>

// ---------------- problem constants ----------------
#define B_    32
#define S_    512
#define H_    512
#define IN_   41
#define OUTD  41
#define T_    (12 * S_)          // 6144 decode steps
#define NGATE 32                 // gate WGs: 16 hidden units each
#define NFC   3                  // fc WGs: 16 output cols each (48 padded >= 41)
#define NWG   (NGATE + NFC)
#define NTHR  256                // 8 waves (wave32)
#define KT    16                 // K = 512 -> 16 k-tiles of 32
#define BH    (B_ * H_)
#define NTILE (KT * 512)         // halves per packed B n-tile

typedef __attribute__((ext_vector_type(16))) _Float16 v16h;
typedef __attribute__((ext_vector_type(8)))  _Float16 v8h;
typedef __attribute__((ext_vector_type(8)))  float    v8f;

__device__ __forceinline__ float sigm(float v) { return 1.f / (1.f + __expf(-v)); }

// ---------------- device-wide sense-reversing barrier ----------------
__device__ __forceinline__ void grid_sync(unsigned* cnt, volatile unsigned* gen) {
  __syncthreads();
  if (threadIdx.x == 0) {
    __threadfence();
    unsigned g = *gen;
    if (atomicAdd(cnt, 1u) == (unsigned)(NWG - 1)) {
      *cnt = 0u;
      __threadfence();
      atomicAdd((unsigned*)gen, 1u);
    } else {
      while (*gen == g) __builtin_amdgcn_s_sleep(2);
    }
    __threadfence();
  }
  __syncthreads();
}

// ---------------- weight packing into WMMA B-operand layout ----------------
// B tile (32K x 16N) per k-tile: lane L holds col N = L&15, K range
// (L<16 ? kt*32+0..15 : kt*32+16..31), 16 contiguous halves per lane.
__device__ __forceinline__ void pack_w(_Float16* dst, const float* src, int rowbase) {
  for (int idx = threadIdx.x; idx < NTILE; idx += NTHR) {
    int kt   = idx >> 9;
    int l    = idx & 511;
    int lane = l >> 4;
    int i    = l & 15;
    int col  = rowbase + (lane & 15);
    int k    = kt * 32 + (lane >> 4) * 16 + i;
    dst[idx] = (_Float16)src[col * H_ + k];
  }
}

// merged r/z weights: W_ih + W_hh (valid because decoder inp_t == h_t for t>=1)
__device__ __forceinline__ void pack_w_sum(_Float16* dst, const float* a,
                                           const float* b, int rowbase) {
  for (int idx = threadIdx.x; idx < NTILE; idx += NTHR) {
    int kt   = idx >> 9;
    int l    = idx & 511;
    int lane = l >> 4;
    int i    = l & 15;
    int col  = rowbase + (lane & 15);
    int k    = kt * 32 + (lane >> 4) * 16 + i;
    dst[idx] = (_Float16)(a[col * H_ + k] + b[col * H_ + k]);
  }
}

__device__ __forceinline__ void pack_wfc(_Float16* dst, const float* Wfc, int fcbase) {
  for (int idx = threadIdx.x; idx < NTILE; idx += NTHR) {
    int kt   = idx >> 9;
    int l    = idx & 511;
    int lane = l >> 4;
    int i    = l & 15;
    int col  = fcbase + (lane & 15);
    int k    = kt * 32 + (lane >> 4) * 16 + i;
    dst[idx] = (col < OUTD) ? (_Float16)Wfc[col * H_ + k] : (_Float16)0.f;
  }
}

// ---------------- one 16x16 C tile: C = A(16x512) * B(512x16), A from LDS ----------------
// A layout (16-bit 16x32): lane<16 row=lane holds K {0..15}; lane>=16 row=lane-16
// holds K {16..31} of each k-tile.
__device__ __forceinline__ void wmma_tile_lds(const _Float16* sH, const _Float16* wb,
                                              float* sC, int mt, int ct, int lane) {
  v8f acc = {};
  int row   = mt * 16 + (lane & 15);
  int abase = (lane & 16) ? 16 : 0;
#pragma unroll
  for (int kt = 0; kt < KT; ++kt) {
    const _Float16* hp = sH + row * H_ + kt * 32 + abase;
    v8h alo = *(const v8h*)hp;
    v8h ahi = *(const v8h*)(hp + 8);
    v16h a;
#pragma unroll
    for (int i = 0; i < 8; ++i) { a[i] = alo[i]; a[8 + i] = ahi[i]; }
    v16h b = *(const v16h*)(wb + kt * 512 + lane * 16);
    acc = __builtin_amdgcn_wmma_f32_16x16x32_f16(false, a, false, b,
                                                 (short)0, acc, false, false);
  }
  // C layout: lane<16 -> M=i,N=lane ; lane>=16 -> M=8+i,N=lane-16
  int M0 = (lane & 16) ? 8 : 0;
  int N  = lane & 15;
  float* cp = sC + ct * 256 + N;
#pragma unroll
  for (int i = 0; i < 8; ++i) cp[(M0 + i) * 16] = acc[i];
}

// Same GEMM but A streamed from global fp16 hidden buffer (fc head).
__device__ __forceinline__ v8f wmma_tile_glb(const _Float16* gH, const _Float16* wb,
                                             int mt, int lane) {
  v8f acc = {};
  int row   = mt * 16 + (lane & 15);
  int abase = (lane & 16) ? 16 : 0;
#pragma unroll
  for (int kt = 0; kt < KT; ++kt) {
    const _Float16* hp = gH + row * H_ + kt * 32 + abase;
    v8h alo = *(const v8h*)hp;
    v8h ahi = *(const v8h*)(hp + 8);
    v16h a;
#pragma unroll
    for (int i = 0; i < 8; ++i) { a[i] = alo[i]; a[8 + i] = ahi[i]; }
    v16h b = *(const v16h*)(wb + kt * 512 + lane * 16);
    acc = __builtin_amdgcn_wmma_f32_16x16x32_f16(false, a, false, b,
                                                 (short)0, acc, false, false);
  }
  return acc;
}

// ---------------- persistent GRU kernel ----------------
__global__ __launch_bounds__(NTHR)
void gru_persistent(const int* __restrict__ x,
                    const float* __restrict__ Wih_e, const float* __restrict__ Whh_e,
                    const float* __restrict__ bih_e, const float* __restrict__ bhh_e,
                    const float* __restrict__ Wih_d, const float* __restrict__ Whh_d,
                    const float* __restrict__ bih_d, const float* __restrict__ bhh_d,
                    const float* __restrict__ Wfc,   const float* __restrict__ bfc,
                    float* __restrict__ out,
                    _Float16* __restrict__ h16,      // [2][B_*H_] (fp16 state exchange)
                    unsigned* sync_cnt, unsigned* sync_gen) {
  // LDS: 4 packed B n-tiles (64KB) + staged h (32KB) + C staging (8KB) + fp32 carry (2KB)
  __shared__ __align__(64) _Float16 sW[4 * NTILE];
  __shared__ __align__(64) _Float16 sH[BH];
  __shared__ __align__(64) float    sC[8 * 256];
  __shared__ float sHP[B_ * 16];               // this WG's fp32 hidden slice (carry)
  __shared__ float sBih[48];
  __shared__ float sBhh[48];

  const int wg   = blockIdx.x;
  const int wave = threadIdx.x >> 5;
  const int lane = threadIdx.x & 31;

  if (wg < NGATE) {
    // ================= gate workgroup: 16 hidden units =================
    const int j0 = wg * 16;

    // zero initial state: global fp16 buffer 0 slice + private fp32 carry
    for (int e = threadIdx.x; e < B_ * 16; e += NTHR) {
      int bb = e >> 4, jj = e & 15;
      h16[bb * H_ + j0 + jj] = (_Float16)0.f;
      sHP[e] = 0.f;
    }
    // pack encoder recurrent weights (3 gate n-tiles) + biases
    for (int nt = 0; nt < 3; ++nt)
      pack_w(sW + nt * NTILE, Whh_e, nt * H_ + j0);
    if (threadIdx.x < 48) {
      int gi = threadIdx.x >> 4, jj = threadIdx.x & 15;
      sBih[threadIdx.x] = bih_e[gi * H_ + j0 + jj];
      sBhh[threadIdx.x] = bhh_e[gi * H_ + j0 + jj];
    }
    __syncthreads();
    grid_sync(sync_cnt, sync_gen);                    // init barrier

    // ---------------- encoder scan: 512 steps ----------------
    for (int s = 0; s < S_; ++s) {
      const int cur = s & 1, nxt = cur ^ 1;
      const _Float16* hsrc = h16 + cur * BH;
      for (int idx = threadIdx.x * 8; idx < BH; idx += NTHR * 8)
        *(v8h*)(sH + idx) = *(const v8h*)(hsrc + idx);
      __syncthreads();
      for (int ct = wave; ct < 6; ct += 8)             // ct = nt*2 + mt, nt in {r,z,n}
        wmma_tile_lds(sH, sW + (ct >> 1) * NTILE, sC, ct & 1, ct, lane);
      __syncthreads();
      for (int e = threadIdx.x; e < B_ * 16; e += NTHR) {
        int bb = e >> 4, jj = e & 15;
        int mt = bb >> 4, mr = bb & 15;
        int jg = j0 + jj;
        int tok = x[bb * S_ + s];                      // one-hot gather
        float xr = Wih_e[(0 * H_ + jg) * IN_ + tok] + sBih[jj];
        float xz = Wih_e[(1 * H_ + jg) * IN_ + tok] + sBih[16 + jj];
        float xn = Wih_e[(2 * H_ + jg) * IN_ + tok] + sBih[32 + jj];
        float hr = sC[(0 * 2 + mt) * 256 + mr * 16 + jj] + sBhh[jj];
        float hz = sC[(1 * 2 + mt) * 256 + mr * 16 + jj] + sBhh[16 + jj];
        float hn = sC[(2 * 2 + mt) * 256 + mr * 16 + jj] + sBhh[32 + jj];
        float r = sigm(xr + hr), z = sigm(xz + hz);
        float n = tanhf(xn + r * hn);
        float hv = (1.f - z) * n + z * sHP[e];
        sHP[e] = hv;
        h16[nxt * BH + bb * H_ + jg] = (_Float16)hv;
      }
      grid_sync(sync_cnt, sync_gen);
    }

    // ---------------- decoder step t=0 (inp==0): hh-only GEMM ----------------
    for (int nt = 0; nt < 3; ++nt)
      pack_w(sW + nt * NTILE, Whh_d, nt * H_ + j0);
    if (threadIdx.x < 48) {
      int gi = threadIdx.x >> 4, jj = threadIdx.x & 15;
      sBih[threadIdx.x] = bih_d[gi * H_ + j0 + jj];
      sBhh[threadIdx.x] = bhh_d[gi * H_ + j0 + jj];
    }
    __syncthreads();
    {
      const _Float16* hsrc = h16;                      // buffer 0 = h_enc
      for (int idx = threadIdx.x * 8; idx < BH; idx += NTHR * 8)
        *(v8h*)(sH + idx) = *(const v8h*)(hsrc + idx);
      __syncthreads();
      for (int ct = wave; ct < 6; ct += 8)
        wmma_tile_lds(sH, sW + (ct >> 1) * NTILE, sC, ct & 1, ct, lane);
      __syncthreads();
      for (int e = threadIdx.x; e < B_ * 16; e += NTHR) {
        int bb = e >> 4, jj = e & 15;
        int mt = bb >> 4, mr = bb & 15;
        int jg = j0 + jj;
        float xr = sBih[jj], xz = sBih[16 + jj], xn = sBih[32 + jj];   // xg = b_ih
        float hr = sC[(0 * 2 + mt) * 256 + mr * 16 + jj] + sBhh[jj];
        float hz = sC[(1 * 2 + mt) * 256 + mr * 16 + jj] + sBhh[16 + jj];
        float hn = sC[(2 * 2 + mt) * 256 + mr * 16 + jj] + sBhh[32 + jj];
        float r = sigm(xr + hr), z = sigm(xz + hz);
        float n = tanhf(xn + r * hn);
        float hv = (1.f - z) * n + z * sHP[e];
        sHP[e] = hv;
        h16[1 * BH + bb * H_ + jg] = (_Float16)hv;
      }
      grid_sync(sync_cnt, sync_gen);
    }

    // ---------------- repack merged decoder weights (4 n-tiles) ----------------
    // nt0 = W_ih_r + W_hh_r, nt1 = W_ih_z + W_hh_z, nt2 = W_ih_n, nt3 = W_hh_n
    pack_w_sum(sW + 0 * NTILE, Wih_d, Whh_d, 0 * H_ + j0);
    pack_w_sum(sW + 1 * NTILE, Wih_d, Whh_d, 1 * H_ + j0);
    pack_w    (sW + 2 * NTILE, Wih_d,        2 * H_ + j0);
    pack_w    (sW + 3 * NTILE, Whh_d,        2 * H_ + j0);
    if (threadIdx.x < 16) {
      int jj = threadIdx.x;
      sBih[jj]      = bih_d[0 * H_ + j0 + jj] + bhh_d[0 * H_ + j0 + jj];  // r bias
      sBih[16 + jj] = bih_d[1 * H_ + j0 + jj] + bhh_d[1 * H_ + j0 + jj];  // z bias
      sBih[32 + jj] = bih_d[2 * H_ + j0 + jj];                            // xn bias
      sBhh[32 + jj] = bhh_d[2 * H_ + j0 + jj];                            // hn bias
    }
    __syncthreads();

    // ---------------- decoder steps 1..T-1: 8 C tiles, one per wave ----------------
    for (int t = 1; t < T_; ++t) {
      const int cur = t & 1, nxt = cur ^ 1;
      const _Float16* hsrc = h16 + cur * BH;
      for (int idx = threadIdx.x * 8; idx < BH; idx += NTHR * 8)
        *(v8h*)(sH + idx) = *(const v8h*)(hsrc + idx);
      __syncthreads();
      {
        int ct = wave;                                 // 8 tiles / 8 waves, ct = nt*2+mt
        wmma_tile_lds(sH, sW + (ct >> 1) * NTILE, sC, ct & 1, ct, lane);
      }
      __syncthreads();
      for (int e = threadIdx.x; e < B_ * 16; e += NTHR) {
        int bb = e >> 4, jj = e & 15;
        int mt = bb >> 4, mr = bb & 15;
        int jg = j0 + jj;
        float rv = sC[(0 * 2 + mt) * 256 + mr * 16 + jj] + sBih[jj];
        float zv = sC[(1 * 2 + mt) * 256 + mr * 16 + jj] + sBih[16 + jj];
        float xn = sC[(2 * 2 + mt) * 256 + mr * 16 + jj] + sBih[32 + jj];
        float hn = sC[(3 * 2 + mt) * 256 + mr * 16 + jj] + sBhh[32 + jj];
        float r = sigm(rv), z = sigm(zv);
        float n = tanhf(xn + r * hn);
        float hv = (1.f - z) * n + z * sHP[e];
        sHP[e] = hv;
        h16[nxt * BH + bb * H_ + jg] = (_Float16)hv;
      }
      grid_sync(sync_cnt, sync_gen);
    }
  } else {
    // ================= fc workgroup: 16 output cols of the H->41 head =================
    const int fcbase = (wg - NGATE) * 16;
    pack_wfc(sW, Wfc, fcbase);
    __syncthreads();
    grid_sync(sync_cnt, sync_gen);                     // init barrier
    for (int s = 0; s < S_; ++s) grid_sync(sync_cnt, sync_gen);  // idle through encoder

    for (int t = 0; t < T_; ++t) {
      if (t >= 1 && wave < 2) {
        // out_{t-1} = h_new(step t-1) @ Wfc^T + bfc ; that buffer is read-only this step
        v8f acc = wmma_tile_glb(h16 + (t & 1) * BH, sW, wave, lane);
        int N = lane & 15, c = fcbase + N;
        if (c < OUTD) {
          int M0 = (lane & 16) ? 8 : 0;
          float bv = bfc[c];
#pragma unroll
          for (int i = 0; i < 8; ++i) {
            int brow = wave * 16 + M0 + i;
            out[((size_t)brow * T_ + (t - 1)) * OUTD + c] = acc[i] + bv;
          }
        }
      }
      grid_sync(sync_cnt, sync_gen);
    }
    // final step's output (h buffer index T_&1 == 0)
    if (wave < 2) {
      v8f acc = wmma_tile_glb(h16 + (T_ & 1) * BH, sW, wave, lane);
      int N = lane & 15, c = fcbase + N;
      if (c < OUTD) {
        int M0 = (lane & 16) ? 8 : 0;
        float bv = bfc[c];
#pragma unroll
        for (int i = 0; i < 8; ++i) {
          int brow = wave * 16 + M0 + i;
          out[((size_t)brow * T_ + (T_ - 1)) * OUTD + c] = acc[i] + bv;
        }
      }
    }
  }
}

// ---------------- host-side launch ----------------
extern "C" void kernel_launch(void* const* d_in, const int* in_sizes, int n_in,
                              void* d_out, int out_size, void* d_ws, size_t ws_size,
                              hipStream_t stream) {
  const int*   x     = (const int*)  d_in[0];
  /* d_in[1] = dayIdx: unused by the reference computation */
  const float* Wih_e = (const float*)d_in[2];
  const float* Whh_e = (const float*)d_in[3];
  const float* bih_e = (const float*)d_in[4];
  const float* bhh_e = (const float*)d_in[5];
  const float* Wih_d = (const float*)d_in[6];
  const float* Whh_d = (const float*)d_in[7];
  const float* bih_d = (const float*)d_in[8];
  const float* bhh_d = (const float*)d_in[9];
  const float* Wfc   = (const float*)d_in[10];
  const float* bfc   = (const float*)d_in[11];

  // workspace layout: [0,256): barrier state | h16[2][BH] fp16
  unsigned char* ws = (unsigned char*)d_ws;
  unsigned* sync_cnt = (unsigned*)ws;
  unsigned* sync_gen = (unsigned*)(ws + 64);
  _Float16* h16 = (_Float16*)(ws + 256);

  hipMemsetAsync(d_ws, 0, 256, stream);   // reset barrier each call (capture-legal)

  gru_persistent<<<dim3(NWG), dim3(NTHR), 0, stream>>>(
      x, Wih_e, Whh_e, bih_e, bhh_e, Wih_d, Whh_d, bih_d, bhh_d, Wfc, bfc,
      (float*)d_out, h16, sync_cnt, sync_gen);
}
